// APPNP_44341242364234
// MI455X (gfx1250) — compile-verified, matrix operand order checked by reading, not run on hardware
//
#include <hip/hip_runtime.h>
#include <hip/hip_bf16.h>

// ---------------------------------------------------------------------------
// APPNP for MI455X (gfx1250, wave32):
//   h0 = relu(x @ W1^T + b1) @ W2^T + b2            (fused WMMA bf16 kernel)
//   10x: h = 0.9 * scatter_add(norm * h[row]) + 0.1*h0   (L2-resident atomics)
//   out = log_softmax(h)
// ---------------------------------------------------------------------------

#define NFEAT 512
#define NHID  64
#define NCLS  16
#define ALPHA_F 0.1f
#define KITERS 10

typedef __attribute__((ext_vector_type(16))) __bf16 v16bf;
typedef __attribute__((ext_vector_type(8)))  float  v8f;

static __device__ inline v16bf pack_bf16(float4 a, float4 b, float4 c, float4 d) {
  v16bf r;
  r[0]  = (__bf16)a.x; r[1]  = (__bf16)a.y; r[2]  = (__bf16)a.z; r[3]  = (__bf16)a.w;
  r[4]  = (__bf16)b.x; r[5]  = (__bf16)b.y; r[6]  = (__bf16)b.z; r[7]  = (__bf16)b.w;
  r[8]  = (__bf16)c.x; r[9]  = (__bf16)c.y; r[10] = (__bf16)c.z; r[11] = (__bf16)c.w;
  r[12] = (__bf16)d.x; r[13] = (__bf16)d.y; r[14] = (__bf16)d.z; r[15] = (__bf16)d.w;
  return r;
}

// Fused MLP: one wave computes a 16-node row tile.
// GEMM1: 16x512 @ 512x64 (4 accumulators of 16x16, 16 K-steps of 32)
// then relu+bias -> LDS transpose stage -> GEMM2: 16x64 @ 64x16 (2 K-steps)
__global__ __launch_bounds__(256) void appnp_mlp_kernel(
    const float* __restrict__ x,  const float* __restrict__ W1,
    const float* __restrict__ b1, const float* __restrict__ W2,
    const float* __restrict__ b2, float* __restrict__ h0, int n)
{
  // 8 waves/block, each wave owns a private [16 rows][64+pad] staging tile.
  __shared__ __align__(16) float lds_h[8][16][NHID + 4];   // pitch 68: conflict-free

  const int wave = threadIdx.x >> 5;
  const int lane = threadIdx.x & 31;
  const int lo   = lane & 15;     // M (A) / N (B,C,D)
  const int hi   = lane >> 4;     // K-half selector (A/B), +8 row selector (C/D)

  const int ntiles = (n + 15) >> 4;
  int tile = blockIdx.x * 8 + wave;
  const bool valid = (tile < ntiles);
  if (!valid) tile = 0;           // keep EXEC all-ones through the WMMA region
  const int m_base = tile << 4;

  v8f acc[4];
#pragma unroll
  for (int t = 0; t < 4; ++t)
#pragma unroll
    for (int r = 0; r < 8; ++r) acc[t][r] = 0.0f;

  const float* xrow = x + (long long)(m_base + lo) * NFEAT;

#pragma unroll 4
  for (int kb = 0; kb < NFEAT; kb += 32) {
    // A fragment: rows of x, bf16-converted. Two contiguous 8-float runs/lane.
    const float4* a01 = (const float4*)(xrow + kb + hi * 8);
    const float4* a23 = (const float4*)(xrow + kb + 16 + hi * 8);
    if (kb + 32 < NFEAT) __builtin_prefetch(xrow + kb + 32, 0, 0);
    v16bf A = pack_bf16(a01[0], a01[1], a23[0], a23[1]);
#pragma unroll
    for (int nt = 0; nt < 4; ++nt) {
      const float* wrow = W1 + (long long)(nt * 16 + lo) * NFEAT + kb + hi * 8;
      const float4* b01 = (const float4*)wrow;
      const float4* b23 = (const float4*)(wrow + 16);
      v16bf B = pack_bf16(b01[0], b01[1], b23[0], b23[1]);
      acc[nt] = __builtin_amdgcn_wmma_f32_16x16x32_bf16(
          false, A, false, B, (short)0, acc[nt], false, false);
    }
  }

  // relu(acc + b1) -> LDS (C/D layout: reg r = row hi*8+r, lane = col)
#pragma unroll
  for (int nt = 0; nt < 4; ++nt) {
    float bias = b1[nt * 16 + lo];
#pragma unroll
    for (int r = 0; r < 8; ++r) {
      float v = acc[nt][r] + bias;
      lds_h[wave][hi * 8 + r][nt * 16 + lo] = v > 0.0f ? v : 0.0f;
    }
  }
  __syncthreads();   // cross-lane visibility for the transpose-read below

  // GEMM2: A = h tile (from LDS, A-layout), B = W2 rows, C init = b2
  v8f acc2;
  {
    float bias2 = b2[lo];
#pragma unroll
    for (int r = 0; r < 8; ++r) acc2[r] = bias2;
  }
#pragma unroll
  for (int kb = 0; kb < NHID; kb += 32) {
    const float* ar = &lds_h[wave][lo][kb + hi * 8];
    float4 a0 = *(const float4*)ar;
    float4 a1 = *(const float4*)(ar + 4);
    const float* ar2 = &lds_h[wave][lo][kb + 16 + hi * 8];
    float4 a2 = *(const float4*)ar2;
    float4 a3 = *(const float4*)(ar2 + 4);
    v16bf A = pack_bf16(a0, a1, a2, a3);

    const float* wrow = W2 + (long long)lo * NHID + kb + hi * 8;
    const float4* b01 = (const float4*)wrow;
    const float4* b23 = (const float4*)(wrow + 16);
    v16bf B = pack_bf16(b01[0], b01[1], b23[0], b23[1]);

    acc2 = __builtin_amdgcn_wmma_f32_16x16x32_bf16(
        false, A, false, B, (short)0, acc2, false, false);
  }

  if (valid) {
#pragma unroll
    for (int r = 0; r < 8; ++r) {
      int row = m_base + hi * 8 + r;
      if (row < n) h0[(long long)row * NCLS + lo] = acc2[r];
    }
  }
}

// ---- propagation helpers ---------------------------------------------------

__global__ __launch_bounds__(256) void deg_one_kernel(unsigned* deg, int n) {
  int v = blockIdx.x * blockDim.x + threadIdx.x;
  if (v < n) deg[v] = 1u;                 // self-loop
}

__global__ __launch_bounds__(256) void deg_count_kernel(
    const long long* __restrict__ col, long long ne, unsigned* __restrict__ deg) {
  long long e = (long long)blockIdx.x * blockDim.x + threadIdx.x;
  if (e < ne) atomicAdd(&deg[(int)col[e]], 1u);
}

__global__ __launch_bounds__(256) void dis_kernel(
    const unsigned* __restrict__ deg, float* __restrict__ dis, int n) {
  int v = blockIdx.x * blockDim.x + threadIdx.x;
  if (v < n) dis[v] = rsqrtf((float)deg[v]);
}

// agg[v] = h[v] * dis[v]^2   (analytic self-loop term; also initializes agg)
__global__ __launch_bounds__(256) void prop_init_kernel(
    const float* __restrict__ h, const float* __restrict__ dis,
    float* __restrict__ agg, long long n16) {
  long long i = (long long)blockIdx.x * blockDim.x + threadIdx.x;
  if (i >= n16) return;
  float d = dis[(int)(i >> 4)];
  agg[i] = h[i] * d * d;
}

// 16 threads per edge (one per class feature) -> coalesced f32 atomics in L2
__global__ __launch_bounds__(256) void prop_scatter_kernel(
    const long long* __restrict__ ei, long long ne,
    const float* __restrict__ h, const float* __restrict__ dis,
    float* __restrict__ agg) {
  long long gid = (long long)blockIdx.x * blockDim.x + threadIdx.x;
  long long e = gid >> 4;
  int f = (int)(gid & 15);
  if (e >= ne) return;
  int r = (int)ei[e];
  int c = (int)ei[ne + e];
  float nrm = dis[r] * dis[c];
  unsafeAtomicAdd(&agg[(long long)c * NCLS + f], nrm * h[(long long)r * NCLS + f]);
}

__global__ __launch_bounds__(256) void prop_combine_kernel(
    const float* __restrict__ agg, const float* __restrict__ h0,
    float* __restrict__ h, long long n16) {
  long long i = (long long)blockIdx.x * blockDim.x + threadIdx.x;
  if (i < n16) h[i] = (1.0f - ALPHA_F) * agg[i] + ALPHA_F * h0[i];
}

__global__ __launch_bounds__(256) void logsoftmax_kernel(
    const float* __restrict__ h, float* __restrict__ out, int n) {
  int v = blockIdx.x * blockDim.x + threadIdx.x;
  if (v >= n) return;
  const float4* hv = (const float4*)(h + (long long)v * NCLS);
  float4 q[4] = { hv[0], hv[1], hv[2], hv[3] };
  float vals[16];
#pragma unroll
  for (int i = 0; i < 4; ++i) {
    vals[4 * i + 0] = q[i].x; vals[4 * i + 1] = q[i].y;
    vals[4 * i + 2] = q[i].z; vals[4 * i + 3] = q[i].w;
  }
  float m = vals[0];
#pragma unroll
  for (int i = 1; i < 16; ++i) m = fmaxf(m, vals[i]);
  float s = 0.0f;
#pragma unroll
  for (int i = 0; i < 16; ++i) s += __expf(vals[i] - m);
  float shift = m + __logf(s);
  float4* ov = (float4*)(out + (long long)v * NCLS);
#pragma unroll
  for (int i = 0; i < 4; ++i) {
    float4 o;
    o.x = vals[4 * i + 0] - shift; o.y = vals[4 * i + 1] - shift;
    o.z = vals[4 * i + 2] - shift; o.w = vals[4 * i + 3] - shift;
    ov[i] = o;
  }
}

// ---------------------------------------------------------------------------

extern "C" void kernel_launch(void* const* d_in, const int* in_sizes, int n_in,
                              void* d_out, int out_size, void* d_ws, size_t ws_size,
                              hipStream_t stream) {
  const float*     x  = (const float*)d_in[0];
  const float*     W1 = (const float*)d_in[1];
  const float*     b1 = (const float*)d_in[2];
  const float*     W2 = (const float*)d_in[3];
  const float*     b2 = (const float*)d_in[4];
  const long long* ei = (const long long*)d_in[5];

  const int       n  = in_sizes[0] / NFEAT;
  const long long ne = (long long)in_sizes[5] / 2;
  float* out = (float*)d_out;

  // workspace layout (all fp32 unless noted): ~20 MB total
  float*    h0  = (float*)d_ws;                 // n*16
  float*    hc  = h0 + (size_t)n * NCLS;        // n*16
  float*    agg = hc + (size_t)n * NCLS;        // n*16
  float*    dis = agg + (size_t)n * NCLS;       // n
  unsigned* deg = (unsigned*)(dis + n);         // n (u32)

  const int TB = 256;
  const int ntiles = (n + 15) / 16;

  appnp_mlp_kernel<<<(ntiles + 7) / 8, TB, 0, stream>>>(x, W1, b1, W2, b2, h0, n);

  deg_one_kernel<<<(n + TB - 1) / TB, TB, 0, stream>>>(deg, n);
  deg_count_kernel<<<(int)((ne + TB - 1) / TB), TB, 0, stream>>>(ei + ne, ne, deg);
  dis_kernel<<<(n + TB - 1) / TB, TB, 0, stream>>>(deg, dis, n);

  const long long n16 = (long long)n * NCLS;
  const int gelem = (int)((n16 + TB - 1) / TB);
  const int gscat = (int)((ne * 16 + TB - 1) / TB);

  const float* hsrc = h0;
  for (int k = 0; k < KITERS; ++k) {
    prop_init_kernel<<<gelem, TB, 0, stream>>>(hsrc, dis, agg, n16);
    prop_scatter_kernel<<<gscat, TB, 0, stream>>>(ei, ne, hsrc, dis, agg);
    prop_combine_kernel<<<gelem, TB, 0, stream>>>(agg, h0, hc, n16);
    hsrc = hc;
  }

  logsoftmax_kernel<<<(n + TB - 1) / TB, TB, 0, stream>>>(hc, out, n);
}
